// Tracks_43310450213418
// MI455X (gfx1250) — matrix operand order repaired and checked by reading, not compiled, not run on hardware
//
#include <hip/hip_runtime.h>
#include <hip/hip_bf16.h>

#define TT 64
#define GG 1024
#define DD 128
#define NN (TT * GG)          // 65536
#define OUTC (3 + DD)         // 131

#define CHUNK_TILES 8                       // B tiles staged per chunk
#define CHUNK_HALVES (CHUNK_TILES * 16 * DD) // 16384 halves = 32 KB
#define NCHUNK (GG / 16 / CHUNK_TILES)      // 8 chunks per time-group sweep

typedef __attribute__((ext_vector_type(16))) _Float16 v16h;
typedef __attribute__((ext_vector_type(8)))  _Float16 v8h;
typedef __attribute__((ext_vector_type(4)))  _Float16 v4h;
typedef __attribute__((ext_vector_type(8)))  float    v8f;

// ---------------------------------------------------------------------------
// 1) Row-normalize features (f32 -> f16), zero visited, seed nxt=-1 for last
//    time group.  One wave per row: 32 lanes x 4 floats = 128 cols.
// ---------------------------------------------------------------------------
__global__ void __launch_bounds__(256)
normalize_kernel(const float* __restrict__ feat,
                 _Float16* __restrict__ fnh,
                 unsigned char* __restrict__ visited,
                 int* __restrict__ nxt) {
    const int wave = threadIdx.x >> 5;
    const int lane = threadIdx.x & 31;
    const int row  = blockIdx.x * 8 + wave;          // grid = NN/8 blocks

    const float4 v = *(const float4*)(feat + (size_t)row * DD + lane * 4);
    float ss = v.x * v.x + v.y * v.y + v.z * v.z + v.w * v.w;
    #pragma unroll
    for (int m = 16; m >= 1; m >>= 1) ss += __shfl_xor(ss, m, 32);
    const float s = 1.0f / fmaxf(sqrtf(ss), 1e-6f);

    v4h h;
    h[0] = (_Float16)(v.x * s);
    h[1] = (_Float16)(v.y * s);
    h[2] = (_Float16)(v.z * s);
    h[3] = (_Float16)(v.w * s);
    *(v4h*)(fnh + (size_t)row * DD + lane * 4) = h;

    if (lane == 0) {
        visited[row] = 0;                    // fresh visited set every call
        if (row >= (TT - 1) * GG) nxt[row] = -1;   // last group has no next
    }
}

// ---------------------------------------------------------------------------
// Fragment loaders for v_wmma_f32_16x16x32_f16 (wave32).
// A (16x32, MxK), per ISA: lanes 0-15 row=lane, K = {0..7, 16..23};
//                          lanes 16-31 row=lane-16, K = {8..15, 24..31}.
// B (32x16, KxN), per ISA: lanes 0-15 col=lane, K = {0..7, 16..23};
//                          lanes 16-31 col=lane-16, K = {8..15, 24..31}.
// ---------------------------------------------------------------------------
__device__ __forceinline__ v16h load_fragA(const _Float16* __restrict__ tile, int lane) {
    const int m  = lane & 15;
    const int kb = (lane >> 4) << 3;                     // 0 or 8
    const _Float16* p = tile + (size_t)m * DD + kb;
    v8h lo = *(const v8h*)(p);                           // K = kb .. kb+7
    v8h hi = *(const v8h*)(p + 16);                      // K = kb+16 .. kb+23
    return __builtin_shufflevector(lo, hi, 0, 1, 2, 3, 4, 5, 6, 7,
                                           8, 9, 10, 11, 12, 13, 14, 15);
}

// B fragment from an LDS-resident tile (row-major 16 x DD) -> ds_load_b128 x2
__device__ __forceinline__ v16h load_fragB_lds(const _Float16* tile, int lane) {
    const int n  = lane & 15;
    const int kb = (lane >> 4) << 3;                     // 0 or 8
    const _Float16* p = tile + n * DD + kb;
    v8h lo = *(const v8h*)(p);                           // K = kb .. kb+7
    v8h hi = *(const v8h*)(p + 16);                      // K = kb+16 .. kb+23
    return __builtin_shufflevector(lo, hi, 0, 1, 2, 3, 4, 5, 6, 7,
                                           8, 9, 10, 11, 12, 13, 14, 15);
}

// ---------------------------------------------------------------------------
// Async stage of one 32 KB chunk (8 B-tiles) into LDS.
// 256 threads x 16 B x 8 issues; tracked with ASYNCcnt (8 per wave).
// ---------------------------------------------------------------------------
__device__ __forceinline__ void stage_chunk(const _Float16* __restrict__ gsrc,
                                            _Float16* lds_dst, int tid) {
    #pragma unroll
    for (int i = 0; i < 8; ++i) {
        const int off = (tid + i * 256) * 8;             // halves (16 B each)
        const unsigned int loff = (unsigned int)(uintptr_t)(lds_dst + off);
        const _Float16* g = gsrc + off;
        asm volatile("global_load_async_to_lds_b128 %0, %1, off"
                     :: "v"(loff), "v"(g)
                     : "memory");
    }
}

// ---------------------------------------------------------------------------
// 2) Fused GEMM + row-argmax with LDS double-buffered B operand:
//    nxt[t*G+g] = (t+1)*G + argmax_h fn[t,g].fn[t+1,h]
//    grid = (TT-1, 8), block = 256 (8 waves, one 16-row A tile per wave).
// ---------------------------------------------------------------------------
__global__ void __launch_bounds__(256)
argmax_kernel(const _Float16* __restrict__ fnh, int* __restrict__ nxt) {
    __shared__ __align__(64) _Float16 smem[2 * CHUNK_HALVES];   // 64 KB

    const int t      = blockIdx.x;                 // 0 .. TT-2
    const int tid    = threadIdx.x;
    const int wave   = tid >> 5;
    const int lane   = tid & 31;
    const int m_tile = blockIdx.y * 8 + wave;      // 0 .. 63
    const int ncol   = lane & 15;

    const _Float16* A  = fnh + ((size_t)t * GG + (size_t)m_tile * 16) * DD;
    const _Float16* B0 = fnh + ((size_t)(t + 1) * GG) * DD;

    // A fragments live in registers for the whole sweep (4 x 8 VGPRs).
    const v16h a0 = load_fragA(A + 0,  lane);
    const v16h a1 = load_fragA(A + 32, lane);
    const v16h a2 = load_fragA(A + 64, lane);
    const v16h a3 = load_fragA(A + 96, lane);

    float bestv[8];
    int   besti[8];
    #pragma unroll
    for (int r = 0; r < 8; ++r) { bestv[r] = -__builtin_inff(); besti[r] = 0; }

    stage_chunk(B0, smem, tid);                    // prime buffer 0

    for (int ch = 0; ch < NCHUNK; ++ch) {
        if (ch + 1 < NCHUNK) {
            // overlap: stage chunk ch+1 while chunk ch is consumed
            stage_chunk(B0 + (size_t)(ch + 1) * CHUNK_HALVES,
                        smem + ((ch + 1) & 1) * CHUNK_HALVES, tid);
            // in-order completion: <=8 outstanding => chunk ch fully in LDS
            asm volatile("s_wait_asynccnt 0x8" ::: "memory");
        } else {
            asm volatile("s_wait_asynccnt 0x0" ::: "memory");
        }
        __syncthreads();                           // all waves' stages visible

        const _Float16* buf = smem + (ch & 1) * CHUNK_HALVES;
        #pragma unroll
        for (int j = 0; j < CHUNK_TILES; ++j) {
            const _Float16* Bt = buf + j * 16 * DD;
            const v16h b0 = load_fragB_lds(Bt + 0,  lane);
            const v16h b1 = load_fragB_lds(Bt + 32, lane);
            const v16h b2 = load_fragB_lds(Bt + 64, lane);
            const v16h b3 = load_fragB_lds(Bt + 96, lane);

            v8f c = {};
            c = __builtin_amdgcn_wmma_f32_16x16x32_f16(false, a0, false, b0, (short)0, c, false, false);
            c = __builtin_amdgcn_wmma_f32_16x16x32_f16(false, a1, false, b1, (short)0, c, false, false);
            c = __builtin_amdgcn_wmma_f32_16x16x32_f16(false, a2, false, b2, (short)0, c, false, false);
            c = __builtin_amdgcn_wmma_f32_16x16x32_f16(false, a3, false, b3, (short)0, c, false, false);

            const int h = (ch * CHUNK_TILES + j) * 16 + ncol;
            #pragma unroll
            for (int r = 0; r < 8; ++r) {
                const float cv = c[r];
                if (cv > bestv[r]) { bestv[r] = cv; besti[r] = h; }
            }
        }
        __syncthreads();                           // protect buffer for reuse
    }

    // Butterfly argmax across the 16 lanes sharing a row set
    // (lanes 0-15 hold M=0..7, lanes 16-31 hold M=8..15).
    #pragma unroll
    for (int m = 8; m >= 1; m >>= 1) {
        #pragma unroll
        for (int r = 0; r < 8; ++r) {
            const float ov = __shfl_xor(bestv[r], m, 16);
            const int   oi = __shfl_xor(besti[r], m, 16);
            if (ov > bestv[r] || (ov == bestv[r] && oi < besti[r])) {
                bestv[r] = ov; besti[r] = oi;      // first-occurrence tie-break
            }
        }
    }

    if (ncol == 0) {
        const int g0 = m_tile * 16 + (lane >> 4) * 8;    // M = r + 8*(lane>=16)
        #pragma unroll
        for (int r = 0; r < 8; ++r)
            nxt[t * GG + g0 + r] = (t + 1) * GG + besti[r];
    }
}

// ---------------------------------------------------------------------------
// 3) Sequential greedy traversal + stable length-filtered compaction.
//    Inherently serial (visited set) -> 1 thread.
// ---------------------------------------------------------------------------
__global__ void traverse_kernel(const int* __restrict__ nxt,
                                int* __restrict__ out_nodes,
                                int* __restrict__ cid,
                                int* __restrict__ clen,
                                unsigned char* __restrict__ visited,
                                int* __restrict__ gather,
                                const int* __restrict__ min_length_p) {
    if (threadIdx.x != 0 || blockIdx.x != 0) return;
    const int min_length = min_length_p[0];

    int p = 0, c = 0;
    for (int start = 0; start < NN; ++start) {
        if (visited[start]) continue;
        visited[start] = 1;
        out_nodes[p] = start; cid[p] = c; ++p;
        int cur = start, len = 1;
        for (;;) {
            const int nx = nxt[cur];
            if (nx < 0 || visited[nx]) break;
            visited[nx] = 1;
            out_nodes[p] = nx; cid[p] = c; ++p;
            cur = nx; ++len;
        }
        clen[c] = len; ++c;
    }
    // stable partition: kept slots first (order preserved), -1 for padding
    int k = 0;
    for (int pp = 0; pp < NN; ++pp)
        if (clen[cid[pp]] >= min_length) gather[k++] = out_nodes[pp];
    for (; k < NN; ++k) gather[k] = -1;
}

// ---------------------------------------------------------------------------
// 4) Parallel gather of [coor || features] rows (zeros for padding).
// ---------------------------------------------------------------------------
__global__ void __launch_bounds__(256)
gather_kernel(const float* __restrict__ coor,
              const float* __restrict__ feat,
              const int* __restrict__ gather,
              float* __restrict__ out) {
    const long long total = (long long)NN * OUTC;
    const long long e = (long long)blockIdx.x * blockDim.x + threadIdx.x;
    if (e >= total) return;
    const int row = (int)(e / OUTC);
    const int col = (int)(e % OUTC);
    const int src = gather[row];
    float v = 0.0f;
    if (src >= 0)
        v = (col < 3) ? coor[(size_t)src * 3 + col]
                      : feat[(size_t)src * DD + (col - 3)];
    out[e] = v;
}

// ---------------------------------------------------------------------------
extern "C" void kernel_launch(void* const* d_in, const int* in_sizes, int n_in,
                              void* d_out, int out_size, void* d_ws, size_t ws_size,
                              hipStream_t stream) {
    const float* coor    = (const float*)d_in[0];   // [NN, 3]
    const float* feat    = (const float*)d_in[1];   // [NN, DD]
    const int*   min_len = (const int*)d_in[2];     // scalar
    float*       out     = (float*)d_out;           // [NN, 131]

    char* ws = (char*)d_ws;
    size_t off = 0;
    _Float16* fnh = (_Float16*)(ws + off); off += (size_t)NN * DD * sizeof(_Float16);
    int* nxt        = (int*)(ws + off); off += (size_t)NN * sizeof(int);
    int* out_nodes  = (int*)(ws + off); off += (size_t)NN * sizeof(int);
    int* cid        = (int*)(ws + off); off += (size_t)NN * sizeof(int);
    int* clen       = (int*)(ws + off); off += (size_t)NN * sizeof(int);
    int* gather     = (int*)(ws + off); off += (size_t)NN * sizeof(int);
    unsigned char* visited = (unsigned char*)(ws + off); off += (size_t)NN;

    normalize_kernel<<<NN / 8, 256, 0, stream>>>(feat, fnh, visited, nxt);
    argmax_kernel<<<dim3(TT - 1, 8), 256, 0, stream>>>(fnh, nxt);
    traverse_kernel<<<1, 1, 0, stream>>>(nxt, out_nodes, cid, clen, visited,
                                         gather, min_len);
    const long long total = (long long)NN * OUTC;
    gather_kernel<<<(int)((total + 255) / 256), 256, 0, stream>>>(coor, feat,
                                                                  gather, out);
}
